// LSTMCellPyTorch_33921651704111
// MI455X (gfx1250) — matrix operand order repaired and checked by reading, not compiled, not run on hardware
//
#include <hip/hip_runtime.h>
#include <hip/hip_bf16.h>

// ---------------------------------------------------------------------------
// LSTM cell, fused: new_c = sigmoid(gf)*c + sigmoid(gi)*tanh(gg)
// gates = [x|h] (16384 x 2048) @ [Wx|Wh]^T (3072 x 2048), K fused to 2048.
// bf16 WMMA (v_wmma_f32_16x16x32_bf16), f32 accumulate, fused epilogue so the
// 192MB f32 gates tensor never exists in memory.
// Round-4 changes:
//  * global loads use uniform char* base + loop-invariant 32-bit BYTE offsets
//    -> backend folds to global_load_b128 saddr+voffset (GVS), no per-stage
//    64-bit VALU address chains.
//  * __builtin_amdgcn_sched_barrier(0) pins the 10-load prefetch clause above
//    the WMMA block so L2 latency drains under the 12 WMMAs instead of being
//    serialized (wait loadcnt 1 + cvt per pair) after them.
// ---------------------------------------------------------------------------

typedef __attribute__((ext_vector_type(16))) __bf16 v16bf;
typedef __attribute__((ext_vector_type(8)))  __bf16 v8bf;
typedef __attribute__((ext_vector_type(4)))  __bf16 v4bf;
typedef __attribute__((ext_vector_type(8)))  float  v8f;
typedef __attribute__((ext_vector_type(4)))  float  f32x4;

#define BATCH   16384
#define HID     1024
#define KDIM    1024          // row length of x, h, and each weight matrix
#define KTOT    2048          // fused K = INPUT + HIDDEN
#define BM      128           // rows (batch) per workgroup
#define BN      64            // cols (hidden) per workgroup, per gate
#define KB      32            // K per stage == WMMA K
#define PITCH   (KB + 8)      // LDS row pitch in bf16 elems (conflict-free frags)
#define A_ELEMS (BM * PITCH)            // 5120
#define B_ELEMS (3 * BN * PITCH)        // 7680
#define STAGE_ELEMS (A_ELEMS + B_ELEMS) // 12800 ushorts = 25600 B / buffer
#define NSTAGES (KTOT / KB)   // 64

__device__ __forceinline__ float sigmoid_f(float v) {
    return 1.0f / (1.0f + __expf(-v));
}
__device__ __forceinline__ float tanh_f(float v) {
    float e = __expf(2.0f * v);        // inf-safe: +inf -> 1, 0 -> -1
    return 1.0f - 2.0f / (e + 1.0f);
}

__global__ __launch_bounds__(256) void lstm_cell_wmma(
    const float* __restrict__ x,   const float* __restrict__ h,
    const float* __restrict__ cin,
    const float* __restrict__ Wxi, const float* __restrict__ Whi,
    const float* __restrict__ Wxf, const float* __restrict__ Whf,
    const float* __restrict__ Wxg, const float* __restrict__ Whg,
    const float* __restrict__ bxi, const float* __restrict__ bhi,
    const float* __restrict__ bxf, const float* __restrict__ bhf,
    const float* __restrict__ bxg, const float* __restrict__ bhg,
    float* __restrict__ out)
{
    __shared__ __align__(16) unsigned short lds[2][STAGE_ELEMS];

    const int tid   = threadIdx.x;
    const int wid   = tid >> 5;     // wave32: 8 waves / block
    const int lane  = tid & 31;
    const int lsub  = lane & 15;
    const int lh    = lane >> 4;    // lane half (ISA fragment layout)
    const int waveM = wid & 3;      // 4 waves along M (32 rows each)
    const int waveN = wid >> 2;     // 2 waves along N (32 cols each)

    const int mBase = blockIdx.x * BM;
    const int nBase = blockIdx.y * BN;

    // Loop-invariant per-thread BYTE offsets for global loads (saddr+voffset
    // GVS addressing) and element offsets for LDS stores.
    unsigned aOff[4], bOff[6];   // global, bytes
    unsigned aSt[4],  bSt[6];    // LDS, ushort elements
#pragma unroll
    for (int j = 0; j < 4; ++j) {
        int id  = j * 256 + tid;
        int row = id >> 3;
        int c4  = (id & 7) << 2;
        aOff[j] = ((unsigned)(mBase + row) * KDIM + c4) * sizeof(float);
        aSt[j]  = (unsigned)row * PITCH + c4;
    }
#pragma unroll
    for (int j = 0; j < 6; ++j) {
        int gate = j >> 1;
        int rem  = (j & 1) * 256 + tid;
        int row  = rem >> 3;
        int c4   = (rem & 7) << 2;
        bOff[j] = ((unsigned)(nBase + row) * KDIM + c4) * sizeof(float);
        bSt[j]  = (unsigned)(A_ELEMS + gate * (BN * PITCH) + row * PITCH + c4);
    }

    f32x4 aReg[4];   // prefetch registers: A tile 128x32 f32 / 256 thr
    f32x4 bReg[6];   // B tiles 3x64x32 f32 / 256 thr

    auto loadRegs = [&](int s) {
        const int  k0 = s * KB;
        const unsigned kkB = (unsigned)(k0 & (KDIM - 1)) * sizeof(float);
        const bool xa = (k0 < KDIM);          // first half: x / Wx, second: h / Wh
        // Uniform stage bases (SALU) + invariant 32-bit byte offsets (GVS form).
        const char* __restrict__ Asrc = (const char*)(xa ? x : h) + kkB;
        const char* __restrict__ Wg0  = (const char*)(xa ? Wxi : Whi) + kkB;
        const char* __restrict__ Wg1  = (const char*)(xa ? Wxf : Whf) + kkB;
        const char* __restrict__ Wg2  = (const char*)(xa ? Wxg : Whg) + kkB;
#pragma unroll
        for (int j = 0; j < 4; ++j)
            aReg[j] = *(const f32x4*)(Asrc + aOff[j]);
#pragma unroll
        for (int j = 0; j < 6; ++j) {
            const int gate = j >> 1;          // compile-time per unrolled j
            const char* __restrict__ Wsrc = gate == 0 ? Wg0 : (gate == 1 ? Wg1 : Wg2);
            bReg[j] = *(const f32x4*)(Wsrc + bOff[j]);
        }
    };

    auto cvtStore = [&](int buf) {
#pragma unroll
        for (int j = 0; j < 4; ++j) {
            // vector fptrunc -> packed v_cvt_pk_bf16_f32 (RNE)
            v4bf cv = __builtin_convertvector(aReg[j], v4bf);
            *(v4bf*)&lds[buf][aSt[j]] = cv;
        }
#pragma unroll
        for (int j = 0; j < 6; ++j) {
            v4bf cv = __builtin_convertvector(bReg[j], v4bf);
            *(v4bf*)&lds[buf][bSt[j]] = cv;
        }
    };

    v8f acc[3][2][2] = {};   // [gate][mi][ni], 96 VGPRs, f32 accumulation

    // Prologue: stage 0 into buffer 0.
    loadRegs(0);
    cvtStore(0);
    __syncthreads();

    for (int s = 0; s < NSTAGES; ++s) {
        const int buf = s & 1;
        // Issue next stage's global loads, then fence the scheduler so they
        // stay hoisted above the WMMA block (latency drains under the WMMAs).
        if (s + 1 < NSTAGES) loadRegs(s + 1);
        __builtin_amdgcn_sched_barrier(0);

        // A fragments: ISA 16-bit A 16x32 layout. Lane half 0 holds K
        // {0..7,16..23}, half 1 holds {8..15,24..31}.
        v16bf aF[2];
#pragma unroll
        for (int mi = 0; mi < 2; ++mi) {
            const unsigned short* pa =
                &lds[buf][(waveM * 32 + mi * 16 + lsub) * PITCH + lh * 8];
            v8bf lo = *(const v8bf*)pa;          // K k+0..7  (half 0) / k+8..15 (half 1)
            v8bf hi = *(const v8bf*)(pa + 16);   // K k+16..23 / k+24..31
            aF[mi] = __builtin_shufflevector(lo, hi,
                       0,1,2,3,4,5,6,7,8,9,10,11,12,13,14,15);
        }
        // B fragments: lane = N column; half 0 holds K {0..7,16..23}, half 1 {8..15,24..31}.
        v16bf bF[3][2];
#pragma unroll
        for (int g = 0; g < 3; ++g) {
#pragma unroll
            for (int ni = 0; ni < 2; ++ni) {
                const unsigned short* pb =
                    &lds[buf][A_ELEMS + g * (BN * PITCH) +
                              (waveN * 32 + ni * 16 + lsub) * PITCH + lh * 8];
                v8bf lo = *(const v8bf*)pb;
                v8bf hi = *(const v8bf*)(pb + 16);
                bF[g][ni] = __builtin_shufflevector(lo, hi,
                              0,1,2,3,4,5,6,7,8,9,10,11,12,13,14,15);
            }
        }

        // ---- 12 back-to-back WMMAs ----
#pragma unroll
        for (int g = 0; g < 3; ++g)
#pragma unroll
            for (int ni = 0; ni < 2; ++ni)
#pragma unroll
                for (int mi = 0; mi < 2; ++mi)
                    acc[g][mi][ni] = __builtin_amdgcn_wmma_f32_16x16x32_bf16(
                        false, aF[mi], false, bF[g][ni],
                        (short)0, acc[g][mi][ni], false, false);

        if (s + 1 < NSTAGES) cvtStore(buf ^ 1);
        __syncthreads();
    }

    // Fused epilogue: biases + sigmoid/tanh + cell update; only new_c hits HBM.
    // C/D layout: VGPR r, lane l: row = 16*tile + r + 8*(l>=16), col = 16*tile + l%16.
#pragma unroll
    for (int ni = 0; ni < 2; ++ni) {
        const int col = nBase + waveN * 32 + ni * 16 + lsub;
        const float bi  = bxi[col] + bhi[col];
        const float bff = bxf[col] + bhf[col];
        const float bg  = bxg[col] + bhg[col];
#pragma unroll
        for (int mi = 0; mi < 2; ++mi) {
#pragma unroll
            for (int r = 0; r < 8; ++r) {
                const int row = mBase + waveM * 32 + mi * 16 + lh * 8 + r;
                const float gi = acc[0][mi][ni][r] + bi;
                const float gf = acc[1][mi][ni][r] + bff;
                const float gg = acc[2][mi][ni][r] + bg;
                const size_t idx = (size_t)row * HID + col;
                out[idx] = sigmoid_f(gf) * cin[idx] + sigmoid_f(gi) * tanh_f(gg);
            }
        }
    }
}

extern "C" void kernel_launch(void* const* d_in, const int* in_sizes, int n_in,
                              void* d_out, int out_size, void* d_ws, size_t ws_size,
                              hipStream_t stream) {
    const float* x   = (const float*)d_in[0];
    const float* h   = (const float*)d_in[1];
    const float* c   = (const float*)d_in[2];
    const float* Wxi = (const float*)d_in[3];
    const float* bxi = (const float*)d_in[4];
    const float* Whi = (const float*)d_in[5];
    const float* bhi = (const float*)d_in[6];
    const float* Wxf = (const float*)d_in[7];
    const float* bxf = (const float*)d_in[8];
    const float* Whf = (const float*)d_in[9];
    const float* bhf = (const float*)d_in[10];
    const float* Wxg = (const float*)d_in[11];
    const float* bxg = (const float*)d_in[12];
    const float* Whg = (const float*)d_in[13];
    const float* bhg = (const float*)d_in[14];
    float* out = (float*)d_out;

    dim3 grid(BATCH / BM, HID / BN);   // 128 x 16 workgroups
    lstm_cell_wmma<<<grid, dim3(256), 0, stream>>>(
        x, h, c, Wxi, Whi, Wxf, Whf, Wxg, Whg,
        bxi, bhi, bxf, bhf, bxg, bhg, out);
}
